// MultiLayerGNN_11768210391116
// MI455X (gfx1250) — matrix operand order repaired and checked by reading, not compiled, not run on hardware
//
#include <hip/hip_runtime.h>
#include <hip/hip_bf16.h>
#include <math.h>

#define NNODES 50000
#define NEDGES 800000
#define DF 128
#define NB 16

typedef __attribute__((ext_vector_type(16))) __bf16 v16bf;
typedef __attribute__((ext_vector_type(8)))  float  v8f;

// -------- float atomic max via monotone bit mapping (works for +/- values) ----
__device__ __forceinline__ void atomicMaxF(float* a, float v) {
    if (v >= 0.f) atomicMax((int*)a, __float_as_int(v));
    else          atomicMin((unsigned int*)a, __float_as_uint(v));
}

// ============================================================================
// Weight pre-pack: W[K x Fout] fp32 -> bf16 fragment-ordered
//   Wp[ks][nslab][lane][j] = bf16( W[(ks*32 + (j>>3)*16 + (lane>>4)*8 + (j&7))*Fout + nslab*16 + (lane&15)] )
// so each lane's 16-element B fragment is one contiguous 32-byte load.
// ============================================================================
__global__ void pack_w(const float* __restrict__ W, __bf16* __restrict__ Wp,
                       int K, int Fout)
{
    long i = (long)blockIdx.x * blockDim.x + threadIdx.x;
    if (i >= (long)K * Fout) return;
    int j = (int)(i & 15);
    long t = i >> 4;
    int lane = (int)(t & 31); t >>= 5;
    int nslabs = Fout >> 4;
    int nslab = (int)(t % nslabs);
    int ks = (int)(t / nslabs);
    int col = lane & 15, hi = lane >> 4;
    Wp[i] = (__bf16)W[(size_t)(ks * 32 + (j >> 3) * 16 + hi * 8 + (j & 7)) * Fout + nslab * 16 + col];
}

__device__ __forceinline__ void cvt_a_frag(v16bf& a, float4 p0, float4 p1,
                                           float4 p2, float4 p3)
{
    a[0]=(__bf16)p0.x; a[1]=(__bf16)p0.y; a[2]=(__bf16)p0.z; a[3]=(__bf16)p0.w;
    a[4]=(__bf16)p1.x; a[5]=(__bf16)p1.y; a[6]=(__bf16)p1.z; a[7]=(__bf16)p1.w;
    a[8]=(__bf16)p2.x; a[9]=(__bf16)p2.y; a[10]=(__bf16)p2.z; a[11]=(__bf16)p2.w;
    a[12]=(__bf16)p3.x; a[13]=(__bf16)p3.y; a[14]=(__bf16)p3.z; a[15]=(__bf16)p3.w;
}

// ============================================================================
// Generic tiled GEMM:  C[M x Fout] = A[M x K] @ W[K x Fout] (+bias) (opt relu)
// bf16 WMMA, fp32 accumulate. One block = 16 rows x Fout cols.
// blockDim.x = 2*Fout (one wave per 16-col slab). Optional per-row domain mask.
// A-frag (16-bit 16x32): lane&15 = M row; per-lane K chunks [k+hi*8,+8) and
// [k+16+hi*8,+8) are contiguous -> float4 loads. B-frag from packed bf16.
// C/D layout: lane&15 = N col; elem i -> M = 8*hi + i
// ============================================================================
__global__ __launch_bounds__(256) void gemm_wmma(
    const float* __restrict__ A, const __bf16* __restrict__ Wp,
    const float* __restrict__ bias, float* __restrict__ C,
    int M, int K, int Fout,
    const int* __restrict__ dom, int dsel, int relu)
{
    const int lane = threadIdx.x & 31;
    const int wave = threadIdx.x >> 5;
    const int n0   = wave * 16;
    const int col  = lane & 15;
    const int hi   = lane >> 4;
    const int rowbase = blockIdx.x * 16;
    const int nslabs  = Fout >> 4;

    int arow = rowbase + col;          // A row carried by this lane
    if (arow >= M) arow = M - 1;
    const float4* A4 = (const float4*)(A + (size_t)arow * K);

    v8f c = {};
    for (int k = 0; k < K; k += 32) {
        const __bf16* wp =
            Wp + ((((size_t)(k >> 5) * nslabs + wave) << 5) + lane) * 16;
        if (k + 32 < K)
            __builtin_prefetch(wp + (size_t)nslabs * 512, 0, 1);
        v16bf b = *(const v16bf*)wp;                      // 32B contiguous
        float4 p0 = A4[(k >> 2) + hi * 2];
        float4 p1 = A4[(k >> 2) + hi * 2 + 1];
        float4 p2 = A4[(k >> 2) + 4 + hi * 2];
        float4 p3 = A4[(k >> 2) + 4 + hi * 2 + 1];
        v16bf a; cvt_a_frag(a, p0, p1, p2, p3);
        c = __builtin_amdgcn_wmma_f32_16x16x32_bf16(false, a, false, b,
                                                    (short)0, c, false, false);
    }
    #pragma unroll
    for (int i = 0; i < 8; ++i) {
        int r = rowbase + hi * 8 + i;
        if (r < M && (dom == nullptr || dom[r] == dsel)) {
            float v = c[i];
            if (bias) v += bias[n0 + col];
            if (relu) v = fmaxf(v, 0.f);
            C[(size_t)r * Fout + n0 + col] = v;
        }
    }
}

// ============================================================================
// Edge-weight MLP (fused):
//   h = relu([x[src] | x[dst] | 1] @ W1 + b1)   (K=257 -> 8 WMMA K-steps + row 256)
//   ew = sigmoid(h @ W2 + b2)
// One block = 16 edges, 8 waves of WMMA, then per-edge reduction from LDS.
// Each 32-wide K block lies entirely in xs (k<128) or xd (k>=128).
// ============================================================================
__global__ __launch_bounds__(256) void edge_mlp(
    const float* __restrict__ X, const int* __restrict__ src,
    const int* __restrict__ dst,
    const __bf16* __restrict__ W1p, const float* __restrict__ W1last,
    const float* __restrict__ b1,
    const float* __restrict__ W2, const float* __restrict__ b2,
    float* __restrict__ ew, int E)
{
    __shared__ float h[16][128];
    const int lane = threadIdx.x & 31;
    const int wave = threadIdx.x >> 5;
    const int n0   = wave * 16;
    const int col  = lane & 15;
    const int hi   = lane >> 4;
    const int ebase = blockIdx.x * 16;

    int e = ebase + col;
    if (e >= E) e = E - 1;
    const float* xs = X + (size_t)src[e] * DF;
    const float* xd = X + (size_t)dst[e] * DF;

    v8f c = {};
    for (int k = 0; k < 256; k += 32) {
        const __bf16* wp = W1p + ((((size_t)(k >> 5) * 8 + wave) << 5) + lane) * 16;
        v16bf b = *(const v16bf*)wp;
        const float4* R4 = (const float4*)((k < DF ? xs : xd) + (k & (DF - 1)));
        float4 p0 = R4[hi * 2];
        float4 p1 = R4[hi * 2 + 1];
        float4 p2 = R4[4 + hi * 2];
        float4 p3 = R4[4 + hi * 2 + 1];
        v16bf a; cvt_a_frag(a, p0, p1, p2, p3);
        c = __builtin_amdgcn_wmma_f32_16x16x32_bf16(false, a, false, b,
                                                    (short)0, c, false, false);
    }
    #pragma unroll
    for (int i = 0; i < 8; ++i) {
        int lr = hi * 8 + i;
        // + constant-1 feature (row 256 of W1) + bias, relu
        float v = c[i] + W1last[n0 + col] + b1[n0 + col];
        h[lr][n0 + col] = fmaxf(v, 0.f);
    }
    __syncthreads();
    if (threadIdx.x < 16) {
        int ee = ebase + threadIdx.x;
        if (ee < E) {
            float acc = b2[0];
            #pragma unroll 4
            for (int f = 0; f < DF; ++f) acc += h[threadIdx.x][f] * W2[f];
            ew[ee] = 1.f / (1.f + expf(-acc));
        }
    }
}

// ---------------- elementwise / utility kernels -----------------------------
__global__ void fill_f(float* p, float v, long n) {
    long i = (long)blockIdx.x * blockDim.x + threadIdx.x;
    if (i < n) p[i] = v;
}

// LayerNorm+relu in place over 128-wide rows, per-domain gamma/beta (wave/row)
__global__ void ln_relu_dom(float* __restrict__ H,
                            const float* __restrict__ g, const float* __restrict__ be,
                            const int* __restrict__ dom, int N)
{
    long t = (long)blockIdx.x * blockDim.x + threadIdx.x;
    int n = (int)(t >> 5), lane = (int)(t & 31);
    if (n >= N) return;
    float* r = H + (size_t)n * DF;
    float s = 0.f;
    for (int f = lane; f < DF; f += 32) s += r[f];
    for (int o = 16; o; o >>= 1) s += __shfl_xor(s, o);
    float mean = s / DF;
    float v = 0.f;
    for (int f = lane; f < DF; f += 32) { float d = r[f] - mean; v += d * d; }
    for (int o = 16; o; o >>= 1) v += __shfl_xor(v, o);
    float rstd = rsqrtf(v / DF + 1e-5f);
    int d = dom[n];
    const float* G = g + (size_t)d * DF;
    const float* B = be + (size_t)d * DF;
    for (int f = lane; f < DF; f += 32)
        r[f] = fmaxf((r[f] - mean) * rstd * G[f] + B[f], 0.f);
}

__global__ void deg_acc(const int* __restrict__ dst, const float* __restrict__ ew,
                        float* __restrict__ deg, int E) {
    int e = blockIdx.x * blockDim.x + threadIdx.x;
    if (e < E) atomicAdd(&deg[dst[e]], ew[e]);
}
__global__ void dinv_k(float* deg, int N) {
    int n = blockIdx.x * blockDim.x + threadIdx.x;
    if (n < N) deg[n] = rsqrtf(deg[n] + 1.0f);   // +1 = self-loop weight; always > 0
}

// wave-per-edge scatter:  out[dst] += h[src] * dinv[src]*ew*dinv[dst]
__global__ void gcn_scatter(const float* __restrict__ h, float* __restrict__ out,
                            const int* __restrict__ src, const int* __restrict__ dst,
                            const float* __restrict__ ew, const float* __restrict__ dinv,
                            int E, int F)
{
    long t = (long)blockIdx.x * blockDim.x + threadIdx.x;
    long e = t >> 5; int lane = (int)(t & 31);
    if (e >= E) return;
    int s = src[e], d = dst[e];
    float nrm = dinv[s] * ew[e] * dinv[d];
    for (int f = lane; f < F; f += 32)
        atomicAdd(&out[(size_t)d * F + f], h[(size_t)s * F + f] * nrm);
}
// out = relu(out + h*dinv^2 + bias)   (self-loop term folded in)
__global__ void gcn_finish(float* __restrict__ out, const float* __restrict__ h,
                           const float* __restrict__ dinv, const float* __restrict__ bias,
                           long n, int F)
{
    long i = (long)blockIdx.x * blockDim.x + threadIdx.x;
    if (i >= n) return;
    int node = (int)(i / F), f = (int)(i % F);
    float dv = dinv[node];
    out[i] = fmaxf(out[i] + h[i] * dv * dv + bias[f], 0.f);
}

// ---------------- GAT pieces ------------------------------------------------
__global__ void gat_scores(const float* __restrict__ h, const float* __restrict__ asrc,
                           const float* __restrict__ adst,
                           float* __restrict__ es, float* __restrict__ ed, int N)
{
    int i = blockIdx.x * blockDim.x + threadIdx.x;
    if (i >= N * 8) return;
    int n = i >> 3, hd = i & 7;
    const float* r = h + (size_t)n * DF + hd * 16;
    float a = 0.f, b = 0.f;
    #pragma unroll
    for (int f = 0; f < 16; ++f) { a += r[f] * asrc[hd * 16 + f]; b += r[f] * adst[hd * 16 + f]; }
    es[i] = a; ed[i] = b;
}
__device__ __forceinline__ void edge_sd(long e, const int* src, const int* dst, int E,
                                        int& s, int& d) {
    if (e < E) { s = src[e]; d = dst[e]; }
    else       { s = (int)(e - E); d = s; }   // appended self-loops
}
__global__ void gat_edge_max(const float* __restrict__ es, const float* __restrict__ ed,
                             const int* __restrict__ src, const int* __restrict__ dst,
                             float* __restrict__ eatt, float* __restrict__ m, int E, int N)
{
    long i = (long)blockIdx.x * blockDim.x + threadIdx.x;
    long EA = (long)E + N;
    if (i >= EA * 8) return;
    long e = i >> 3; int hd = (int)(i & 7);
    int s, d; edge_sd(e, src, dst, E, s, d);
    float v = es[(size_t)s * 8 + hd] + ed[(size_t)d * 8 + hd];
    v = v > 0.f ? v : 0.2f * v;                 // leaky_relu(0.2)
    eatt[i] = v;
    atomicMaxF(&m[(size_t)d * 8 + hd], v);
}
__global__ void m_fix(float* m, long n) {       // where(isfinite(m), m, 0)
    long i = (long)blockIdx.x * blockDim.x + threadIdx.x;
    if (i < n) { float v = m[i]; if (!(v > -3.0e38f && v < 3.0e38f)) m[i] = 0.f; }
}
__global__ void gat_edge_exp(float* __restrict__ eatt, const float* __restrict__ m,
                             float* __restrict__ ssum,
                             const int* __restrict__ src, const int* __restrict__ dst,
                             int E, int N)
{
    long i = (long)blockIdx.x * blockDim.x + threadIdx.x;
    long EA = (long)E + N;
    if (i >= EA * 8) return;
    long e = i >> 3; int hd = (int)(i & 7);
    int s, d; edge_sd(e, src, dst, E, s, d); (void)s;
    float ex = expf(eatt[i] - m[(size_t)d * 8 + hd]);
    eatt[i] = ex;
    atomicAdd(&ssum[(size_t)d * 8 + hd], ex);
}
__global__ void gat_scatter(const float* __restrict__ h, float* __restrict__ out,
                            const float* __restrict__ eatt, const float* __restrict__ ssum,
                            const int* __restrict__ src, const int* __restrict__ dst,
                            int E, int N)
{
    long t = (long)blockIdx.x * blockDim.x + threadIdx.x;
    long e = t >> 5; int lane = (int)(t & 31);
    long EA = (long)E + N;
    if (e >= EA) return;
    int s, d; edge_sd(e, src, dst, E, s, d);
    for (int f = lane; f < DF; f += 32) {
        int hd = f >> 4;
        float alpha = eatt[e * 8 + hd] / (ssum[(size_t)d * 8 + hd] + 1e-16f);
        atomicAdd(&out[(size_t)d * DF + f], h[(size_t)s * DF + f] * alpha);
    }
}
__global__ void add_bias(float* out, const float* b, long n, int F) {
    long i = (long)blockIdx.x * blockDim.x + threadIdx.x;
    if (i < n) out[i] += b[i % F];
}

// ---------------- pooling ---------------------------------------------------
__global__ void pool_cnt(const int* __restrict__ batch, float* __restrict__ cnt, int N) {
    int n = blockIdx.x * blockDim.x + threadIdx.x;
    if (n < N) atomicAdd(&cnt[batch[n]], 1.0f);
}
__global__ void pool_acc(const float* __restrict__ x2, const float* __restrict__ x3,
                         const float* __restrict__ x4, const int* __restrict__ batch,
                         float* __restrict__ p2, float* __restrict__ p3,
                         float* __restrict__ p4, int N)
{
    long t = (long)blockIdx.x * blockDim.x + threadIdx.x;
    int n = (int)(t >> 5), lane = (int)(t & 31);
    if (n >= N) return;
    int b = batch[n];
    for (int f = lane; f < 128; f += 32) atomicAdd(&p2[b * 128 + f], x2[(size_t)n * 128 + f]);
    for (int f = lane; f < 64;  f += 32) atomicAdd(&p3[b * 64 + f],  x3[(size_t)n * 64 + f]);
    for (int f = lane; f < 32;  f += 32) atomicMaxF(&p4[b * 32 + f], x4[(size_t)n * 32 + f]);
}

// ---------------- dense head (tiny: 16 rows) --------------------------------
__global__ __launch_bounds__(256) void head_kernel(
    const float* P2, const float* P3, const float* P4, const float* CNT,
    const float* W1, const float* b1, const float* g1, const float* be1,
    const float* W2, const float* b2, const float* g2, const float* be2,
    const float* W3, const float* b3, float* out)
{
    __shared__ float G[16 * 224];
    __shared__ float H1[16 * 256];
    __shared__ float H2[16 * 128];
    __shared__ float MV[32];
    int tid = threadIdx.x;

    for (int i = tid; i < 16 * 224; i += 256) {
        int b = i / 224, f = i % 224;
        float c = fmaxf(CNT[b], 1.f);
        G[i] = f < 128 ? P2[b * 128 + f] / c
             : f < 192 ? P3[b * 64 + (f - 128)] / c
                       : P4[b * 32 + (f - 192)];
    }
    __syncthreads();
    for (int i = tid; i < 16 * 256; i += 256) {
        int b = i / 256, o = i % 256;
        float acc = b1[o];
        for (int k = 0; k < 224; ++k) acc += G[b * 224 + k] * W1[k * 256 + o];
        H1[i] = acc;
    }
    __syncthreads();
    if (tid < 16) {
        float m = 0.f; for (int o = 0; o < 256; ++o) m += H1[tid * 256 + o]; m /= 256.f;
        float v = 0.f; for (int o = 0; o < 256; ++o) { float d = H1[tid * 256 + o] - m; v += d * d; }
        MV[tid] = m; MV[16 + tid] = rsqrtf(v / 256.f + 1e-5f);
    }
    __syncthreads();
    for (int i = tid; i < 16 * 256; i += 256) {
        int b = i / 256, o = i % 256;
        H1[i] = fmaxf((H1[i] - MV[b]) * MV[16 + b] * g1[o] + be1[o], 0.f);
    }
    __syncthreads();
    for (int i = tid; i < 16 * 128; i += 256) {
        int b = i / 128, o = i % 128;
        float acc = b2[o];
        for (int k = 0; k < 256; ++k) acc += H1[b * 256 + k] * W2[k * 128 + o];
        H2[i] = acc;
    }
    __syncthreads();
    if (tid < 16) {
        float m = 0.f; for (int o = 0; o < 128; ++o) m += H2[tid * 128 + o]; m /= 128.f;
        float v = 0.f; for (int o = 0; o < 128; ++o) { float d = H2[tid * 128 + o] - m; v += d * d; }
        MV[tid] = m; MV[16 + tid] = rsqrtf(v / 128.f + 1e-5f);
    }
    __syncthreads();
    for (int i = tid; i < 16 * 128; i += 256) {
        int b = i / 128, o = i % 128;
        H2[i] = fmaxf((H2[i] - MV[b]) * MV[16 + b] * g2[o] + be2[o], 0.f);
    }
    __syncthreads();
    for (int i = tid; i < 16 * 64; i += 256) {
        int b = i / 64, o = i % 64;
        float acc = b3[o];
        for (int k = 0; k < 128; ++k) acc += H2[b * 128 + k] * W3[k * 64 + o];
        out[i] = acc;
    }
}

// ============================================================================
extern "C" void kernel_launch(void* const* d_in, const int* in_sizes, int n_in,
                              void* d_out, int out_size, void* d_ws, size_t ws_size,
                              hipStream_t stream)
{
    const float* x      = (const float*)d_in[0];
    const int*   ei     = (const int*)d_in[1];
    const int*   batch  = (const int*)d_in[2];
    const int*   dom    = (const int*)d_in[3];
    const float* de_W1  = (const float*)d_in[4];
    const float* de_b1  = (const float*)d_in[5];
    const float* de_g1  = (const float*)d_in[6];
    const float* de_be1 = (const float*)d_in[7];
    const float* de_W2  = (const float*)d_in[8];
    const float* de_b2  = (const float*)d_in[9];
    const float* ew_W1  = (const float*)d_in[10];
    const float* ew_b1  = (const float*)d_in[11];
    const float* ew_W2  = (const float*)d_in[12];
    const float* ew_b2  = (const float*)d_in[13];
    const float* gcn1_W = (const float*)d_in[14];
    const float* gcn1_b = (const float*)d_in[15];
    const float* gcn2_W = (const float*)d_in[16];
    const float* gcn2_b = (const float*)d_in[17];
    const float* gat1_W = (const float*)d_in[18];
    const float* gat1_as= (const float*)d_in[19];
    const float* gat1_ad= (const float*)d_in[20];
    const float* gat1_b = (const float*)d_in[21];
    const float* gat2_W = (const float*)d_in[22];
    const float* gat2_as= (const float*)d_in[23];
    const float* gat2_ad= (const float*)d_in[24];
    const float* gat2_b = (const float*)d_in[25];
    const float* pc1_W  = (const float*)d_in[26];
    const float* pc1_b  = (const float*)d_in[27];
    const float* pc2_W  = (const float*)d_in[28];
    const float* pc2_b  = (const float*)d_in[29];
    const float* d1_W   = (const float*)d_in[30];
    const float* d1_b   = (const float*)d_in[31];
    const float* d1_g   = (const float*)d_in[32];
    const float* d1_be  = (const float*)d_in[33];
    const float* d2_W   = (const float*)d_in[34];
    const float* d2_b   = (const float*)d_in[35];
    const float* d2_g   = (const float*)d_in[36];
    const float* d2_be  = (const float*)d_in[37];
    const float* d3_W   = (const float*)d_in[38];
    const float* d3_b   = (const float*)d_in[39];

    const int* src = ei;
    const int* dst = ei + NEDGES;

    // -------- workspace layout (floats); fits in the 192MB L2 ---------------
    float* ws = (float*)d_ws;
    size_t oA  = 0;
    size_t oT  = oA + (size_t)NNODES * DF;
    size_t oU  = oT + (size_t)NNODES * DF;
    size_t oEW = oU + (size_t)NNODES * DF;
    size_t oDEG= oEW + NEDGES;
    size_t oES = oDEG + NNODES;
    size_t oED = oES + (size_t)NNODES * 8;
    size_t oM  = oED + (size_t)NNODES * 8;
    size_t oS  = oM  + (size_t)NNODES * 8;
    size_t oEATT = oS + (size_t)NNODES * 8;
    size_t oP2 = oEATT + ((size_t)NEDGES + NNODES) * 8;
    size_t oP3 = oP2 + 16 * 128;
    size_t oP4 = oP3 + 16 * 64;
    size_t oCNT= oP4 + 16 * 32;
    size_t oPACK = (oCNT + 16 + 15) & ~(size_t)15;   // 32B-aligned bf16 region
    float *A = ws + oA, *T = ws + oT, *U = ws + oU, *EW = ws + oEW;
    float *DEG = ws + oDEG, *ES = ws + oES, *ED = ws + oED, *M = ws + oM, *S = ws + oS;
    float *EATT = ws + oEATT, *P2 = ws + oP2, *P3 = ws + oP3, *P4 = ws + oP4, *CNT = ws + oCNT;

    // packed bf16 weights (fragment-ordered), tiny (~740 KB total)
    __bf16* PK = (__bf16*)(ws + oPACK);
    __bf16* pk_deW1 = PK;                          // 8 * 128*128
    __bf16* pk_deW2 = pk_deW1 + (size_t)8 * DF * DF;
    __bf16* pk_ew   = pk_deW2 + (size_t)8 * DF * DF;   // 256*128 (row 256 separate)
    __bf16* pk_gcn1 = pk_ew   + (size_t)256 * DF;
    __bf16* pk_gcn2 = pk_gcn1 + (size_t)DF * DF;
    __bf16* pk_gat1 = pk_gcn2 + (size_t)DF * DF;
    __bf16* pk_gat2 = pk_gat1 + (size_t)DF * DF;
    __bf16* pk_pc1  = pk_gat2 + (size_t)DF * DF;       // 128*64
    __bf16* pk_pc2  = pk_pc1  + (size_t)DF * 64;       // 64*32

    const int gN16 = (NNODES + 15) / 16;
    auto blk = [](long n) { return (unsigned)((n + 255) / 256); };
    const long EA  = (long)NEDGES + NNODES;

    // ---- 0. pre-pack all weights to bf16 fragment order ---------------------
    for (int d = 0; d < 8; ++d) {
        pack_w<<<blk(DF * DF), 256, 0, stream>>>(de_W1 + (size_t)d * DF * DF,
                                                 pk_deW1 + (size_t)d * DF * DF, DF, DF);
        pack_w<<<blk(DF * DF), 256, 0, stream>>>(de_W2 + (size_t)d * DF * DF,
                                                 pk_deW2 + (size_t)d * DF * DF, DF, DF);
    }
    pack_w<<<blk(256 * DF), 256, 0, stream>>>(ew_W1, pk_ew, 256, DF);
    pack_w<<<blk(DF * DF), 256, 0, stream>>>(gcn1_W, pk_gcn1, DF, DF);
    pack_w<<<blk(DF * DF), 256, 0, stream>>>(gcn2_W, pk_gcn2, DF, DF);
    pack_w<<<blk(DF * DF), 256, 0, stream>>>(gat1_W, pk_gat1, DF, DF);
    pack_w<<<blk(DF * DF), 256, 0, stream>>>(gat2_W, pk_gat2, DF, DF);
    pack_w<<<blk(DF * 64), 256, 0, stream>>>(pc1_W, pk_pc1, DF, 64);
    pack_w<<<blk(64 * 32), 256, 0, stream>>>(pc2_W, pk_pc2, 64, 32);

    // ---- 1. domain encoders: T = x@W1[d]+b1 (masked), LN+relu, A = T@W2[d]+b2
    for (int d = 0; d < 8; ++d)
        gemm_wmma<<<gN16, 256, 0, stream>>>(x, pk_deW1 + (size_t)d * DF * DF,
                                            de_b1 + d * DF, T, NNODES, DF, DF, dom, d, 0);
    ln_relu_dom<<<blk((long)NNODES * 32), 256, 0, stream>>>(T, de_g1, de_be1, dom, NNODES);
    for (int d = 0; d < 8; ++d)
        gemm_wmma<<<gN16, 256, 0, stream>>>(T, pk_deW2 + (size_t)d * DF * DF,
                                            de_b2 + d * DF, A, NNODES, DF, DF, dom, d, 0);

    // ---- 2. edge-weight MLP (the 52-GFLOP WMMA GEMM) ------------------------
    edge_mlp<<<(NEDGES + 15) / 16, 256, 0, stream>>>(A, src, dst, pk_ew,
                                                     ew_W1 + (size_t)256 * DF, ew_b1,
                                                     ew_W2, ew_b2, EW, NEDGES);

    // ---- 3. degree / dinv ---------------------------------------------------
    fill_f<<<blk(NNODES), 256, 0, stream>>>(DEG, 0.f, NNODES);
    deg_acc<<<blk(NEDGES), 256, 0, stream>>>(dst, EW, DEG, NEDGES);
    dinv_k<<<blk(NNODES), 256, 0, stream>>>(DEG, NNODES);

    auto run_gcn = [&](const float* in, float* out, const __bf16* Wp, const float* bt,
                       int Fin, int Fout) {
        gemm_wmma<<<gN16, Fout * 2, 0, stream>>>(in, Wp, nullptr, T, NNODES, Fin, Fout,
                                                 nullptr, 0, 0);
        fill_f<<<blk((long)NNODES * Fout), 256, 0, stream>>>(out, 0.f, (long)NNODES * Fout);
        gcn_scatter<<<blk((long)NEDGES * 32), 256, 0, stream>>>(T, out, src, dst, EW, DEG,
                                                                NEDGES, Fout);
        gcn_finish<<<blk((long)NNODES * Fout), 256, 0, stream>>>(out, T, DEG, bt,
                                                                 (long)NNODES * Fout, Fout);
    };
    auto run_gat = [&](const float* in, float* out, const __bf16* Wp, const float* as,
                       const float* ad, const float* bt) {
        gemm_wmma<<<gN16, 256, 0, stream>>>(in, Wp, nullptr, T, NNODES, DF, DF, nullptr, 0, 0);
        gat_scores<<<blk((long)NNODES * 8), 256, 0, stream>>>(T, as, ad, ES, ED, NNODES);
        fill_f<<<blk((long)NNODES * 8), 256, 0, stream>>>(M, -INFINITY, (long)NNODES * 8);
        fill_f<<<blk((long)NNODES * 8), 256, 0, stream>>>(S, 0.f, (long)NNODES * 8);
        gat_edge_max<<<blk(EA * 8), 256, 0, stream>>>(ES, ED, src, dst, EATT, M, NEDGES, NNODES);
        m_fix<<<blk((long)NNODES * 8), 256, 0, stream>>>(M, (long)NNODES * 8);
        gat_edge_exp<<<blk(EA * 8), 256, 0, stream>>>(EATT, M, S, src, dst, NEDGES, NNODES);
        fill_f<<<blk((long)NNODES * DF), 256, 0, stream>>>(out, 0.f, (long)NNODES * DF);
        gat_scatter<<<blk(EA * 32), 256, 0, stream>>>(T, out, EATT, S, src, dst, NEDGES, NNODES);
        add_bias<<<blk((long)NNODES * DF), 256, 0, stream>>>(out, bt, (long)NNODES * DF, DF);
    };

    // ---- 4. GCN x2, GAT x2, pooled-GCN x2 ----------------------------------
    run_gcn(A, U, pk_gcn1, gcn1_b, 128, 128);     // U = relu(gcn1(enc))
    run_gcn(U, A, pk_gcn2, gcn2_b, 128, 128);     // A = x1
    run_gat(A, U, pk_gat1, gat1_as, gat1_ad, gat1_b);  // U = gat1(x1)
    run_gat(U, A, pk_gat2, gat2_as, gat2_ad, gat2_b);  // A = x2 (kept for pooling)
    run_gcn(A, U, pk_pc1, pc1_b, 128, 64);        // U[:N*64] = x3
    float* X4 = U + (size_t)NNODES * 64;
    run_gcn(U, X4, pk_pc2, pc2_b, 64, 32);        // X4 = x4

    // ---- 5. pooling + head -------------------------------------------------
    fill_f<<<blk(16 * 128), 256, 0, stream>>>(P2, 0.f, 16 * 128);
    fill_f<<<blk(16 * 64),  256, 0, stream>>>(P3, 0.f, 16 * 64);
    fill_f<<<blk(16 * 32),  256, 0, stream>>>(P4, -INFINITY, 16 * 32);
    fill_f<<<1, 256, 0, stream>>>(CNT, 0.f, 16);
    pool_cnt<<<blk(NNODES), 256, 0, stream>>>(batch, CNT, NNODES);
    pool_acc<<<blk((long)NNODES * 32), 256, 0, stream>>>(A, U, X4, batch, P2, P3, P4, NNODES);
    head_kernel<<<1, 256, 0, stream>>>(P2, P3, P4, CNT,
                                       d1_W, d1_b, d1_g, d1_be,
                                       d2_W, d2_b, d2_g, d2_be,
                                       d3_W, d3_b, (float*)d_out);
}
